// NeighAttnMat_76647986364462
// MI455X (gfx1250) — matrix operand order repaired and checked by reading, not compiled, not run on hardware
//
#include <hip/hip_runtime.h>

// ---------------------------------------------------------------------------
// NeighAttn for MI455X (gfx1250):
//   1) cvt_x   : x fp32 -> f16                       (268MB -> 134MB)
//   2) cvt_w   : W_qk fp32 (512x1024) -> f16 W^T (1024x512), K-contiguous
//   3) gemm_qk : qk = x @ W_qk via v_wmma_f32_16x16x32_f16, fp32 accum
//                64x64 tile per wave (4x4 WMMA grid) -> 32 FLOP/byte from L2
//   4) neigh_attn: 49 neighborhood dots per pixel/head, LDS-tiled halo
// ---------------------------------------------------------------------------

typedef __attribute__((ext_vector_type(16))) _Float16 v16h;
typedef __attribute__((ext_vector_type(8)))  _Float16 h8;
typedef __attribute__((ext_vector_type(4)))  _Float16 h4;
typedef __attribute__((ext_vector_type(8)))  float    v8f;

#define MB    8
#define HH    128
#define WW    128
#define DIM   512
#define ND    1024            // 2*DIM
#define NHEAD 8
#define HD    64
#define KW    7
#define RR    3
#define MTOT  (MB*HH*WW)      // 131072 rows in the GEMM

union AF { v16h v; h8 h[2]; };

// ---------------------------------------------------------------- cvt_x ----
__global__ __launch_bounds__(256)
void cvt_x_kernel(const float* __restrict__ x, _Float16* __restrict__ xh, int n) {
  int t = blockIdx.x * 256 + threadIdx.x;
  int base = t * 8;
  if (base >= n) return;
  float4 a = *(const float4*)(x + base);
  float4 b = *(const float4*)(x + base + 4);
  h8 o = { (_Float16)a.x, (_Float16)a.y, (_Float16)a.z, (_Float16)a.w,
           (_Float16)b.x, (_Float16)b.y, (_Float16)b.z, (_Float16)b.w };
  *(h8*)(xh + base) = o;
}

// ---------------------------------------------------------------- cvt_w ----
// W (DIM x ND, N contiguous) -> Wt f16 (ND x DIM, K contiguous)
__global__ __launch_bounds__(256)
void cvt_w_kernel(const float* __restrict__ w, _Float16* __restrict__ wt) {
  __shared__ float tile[16][17];
  int tx = threadIdx.x & 15, ty = threadIdx.x >> 4;
  int n0 = (blockIdx.x & 63) << 4;   // ND/16 = 64 tiles
  int k0 = (blockIdx.x >> 6) << 4;   // DIM/16 = 32 tiles
  tile[ty][tx] = w[(size_t)(k0 + ty) * ND + n0 + tx];
  __syncthreads();
  wt[(size_t)(n0 + ty) * DIM + k0 + tx] = (_Float16)tile[tx][ty];
}

// --------------------------------------------------------------- gemm ------
// Each wave computes a 64x64 tile of qk (4x4 of 16x16).  Block = 4 waves in
// a 2x2 arrangement covering a 128x128 block tile.  K-loop in steps of 32:
// 8 A-frag loads + 8 B-frag loads feed 16 WMMAs (4x operand reuse).
__global__ __launch_bounds__(128)
void gemm_qk_kernel(const _Float16* __restrict__ xh,
                    const _Float16* __restrict__ wt,
                    float* __restrict__ qk) {
  const int lane = threadIdx.x & 31;
  const int wave = threadIdx.x >> 5;
  const int tilesN = ND / 128;                      // 8
  const int bidN = blockIdx.x % tilesN;
  const int bidM = blockIdx.x / tilesN;
  const int m0 = bidM * 128 + (wave >> 1) * 64;
  const int n0 = bidN * 128 + (wave & 1) * 64;
  const int lm = lane & 15;
  const int lh = lane >> 4;                         // half-wave id (0/1)

  v8f c[4][4] = {};

  // A fragment (16x32 f16): lane holds row m0+lm, K-chunks at 8*lh and 8*lh+16
  const _Float16* aBase = xh + (size_t)(m0 + lm) * DIM + 8 * lh;
  // B fragment (32x16 f16): lane holds col n0+lm, K-chunks at 8*lh and 8*lh+16
  // (Wt is K-contiguous per output column)
  const _Float16* bBase = wt + (size_t)(n0 + lm) * DIM + 8 * lh;

  for (int kk = 0; kk < DIM; kk += 32) {
    AF a[4], b[4];
#pragma unroll
    for (int t = 0; t < 4; ++t) {
      const _Float16* pa = aBase + (size_t)t * 16 * DIM + kk;
      a[t].h[0] = *(const h8*)(pa);
      a[t].h[1] = *(const h8*)(pa + 16);
      const _Float16* pb = bBase + (size_t)t * 16 * DIM + kk;
      b[t].h[0] = *(const h8*)(pb);
      b[t].h[1] = *(const h8*)(pb + 16);
      if (kk + 32 < DIM) {                  // pull next K-slice toward WGP
        __builtin_prefetch(pa + 32, 0, 3);  // -> global_prefetch_b8
        __builtin_prefetch(pb + 32, 0, 3);
      }
    }
#pragma unroll
    for (int i = 0; i < 4; ++i)
#pragma unroll
      for (int j = 0; j < 4; ++j)
        c[i][j] = __builtin_amdgcn_wmma_f32_16x16x32_f16(
            false, a[i].v, false, b[j].v, (short)0, c[i][j], false, false);
  }

  // C layout: VGPR r, lane -> (row m0+i*16 + r + 8*lh, col n0+j*16 + lm)
#pragma unroll
  for (int i = 0; i < 4; ++i)
#pragma unroll
    for (int r = 0; r < 8; ++r) {
      const int row = m0 + i * 16 + r + 8 * lh;
      float* qrow = qk + (size_t)row * ND + n0 + lm;
#pragma unroll
      for (int j = 0; j < 4; ++j)
        qrow[j * 16] = c[i][j][r];
    }
}

// --------------------------------------------------------- neigh_attn ------
// Block = one (batch, head, 16x16 pixel tile).  k halo (22x22 pixels, 64 d)
// staged in LDS as f16 laid out [d4][li][lj] (conflict-free across lanes).
#define TS   16
#define HALO 22

__global__ __launch_bounds__(256)
void neigh_attn_kernel(const float* __restrict__ qk, float* __restrict__ out) {
  __shared__ h4 kt[16][HALO][HALO];    // 16*22*22*8B = 61,952 B

  const int tid = threadIdx.x;
  const int tx = tid & 15, ty = tid >> 4;
  int tmp = blockIdx.x;
  const int tj = tmp & 7; tmp >>= 3;
  const int ti = tmp & 7; tmp >>= 3;
  const int h  = tmp & 7; tmp >>= 3;
  const int b  = tmp;
  const int i0 = ti * TS, j0 = tj * TS;
  const int r0 = (i0 - RR) > 0 ? (i0 - RR) : 0;
  const int c0 = (j0 - RR) > 0 ? (j0 - RR) : 0;
  const size_t imgBase = (size_t)b * (HH * WW);

  // cooperative halo load: k vector lives at qk[p*ND + 512 + h*64 + d]
  for (int idx = tid; idx < 16 * HALO * HALO; idx += 256) {
    int d4   = idx & 15;
    int rest = idx >> 4;
    int li   = rest / HALO;
    int lj   = rest - li * HALO;
    int gi = r0 + li, gj = c0 + lj;
    float4 v = make_float4(0.f, 0.f, 0.f, 0.f);
    if (gi < HH && gj < WW) {
      size_t p = imgBase + (size_t)gi * WW + gj;
      v = *(const float4*)(qk + p * ND + DIM + h * HD + d4 * 4);
    }
    kt[d4][li][lj] = h4{ (_Float16)v.x, (_Float16)v.y,
                         (_Float16)v.z, (_Float16)v.w };
  }
  __syncthreads();

  const int i = i0 + ty, j = j0 + tx;
  const size_t p = imgBase + (size_t)i * WW + j;
  const float scale = 0.125f;                    // HEAD_DIM^-0.5
  const float* qp = qk + p * ND + h * HD;
  float4 q4[16];
#pragma unroll
  for (int d4 = 0; d4 < 16; ++d4) {
    float4 v = *(const float4*)(qp + d4 * 4);
    q4[d4] = make_float4(v.x * scale, v.y * scale, v.z * scale, v.w * scale);
  }

  // window start: clip(i-R, 0, H-K), local to halo tile
  int si = i - RR; si = si < 0 ? 0 : si; si = si > (HH - KW) ? (HH - KW) : si;
  int sj = j - RR; sj = sj < 0 ? 0 : sj; sj = sj > (WW - KW) ? (WW - KW) : sj;
  si -= r0; sj -= c0;

  float* op = out + ((((size_t)b * NHEAD + h) * HH + i) * WW + j) * (KW * KW);
  for (int ki = 0; ki < KW; ++ki) {
#pragma unroll
    for (int kj = 0; kj < KW; ++kj) {
      const int li = si + ki, lj = sj + kj;
      float acc = 0.f;
#pragma unroll
      for (int d4 = 0; d4 < 16; ++d4) {
        h4 kv = kt[d4][li][lj];
        float4 qv = q4[d4];
        acc += qv.x * (float)kv.x;   // clang -> v_fma_mix_f32
        acc += qv.y * (float)kv.y;
        acc += qv.z * (float)kv.z;
        acc += qv.w * (float)kv.w;
      }
      op[ki * KW + kj] = acc;
    }
  }
}

// ------------------------------------------------------------- launch ------
extern "C" void kernel_launch(void* const* d_in, const int* in_sizes, int n_in,
                              void* d_out, int out_size, void* d_ws, size_t ws_size,
                              hipStream_t stream) {
  const float* x   = (const float*)d_in[0];   // (8,128,128,512) fp32
  const float* wqk = (const float*)d_in[1];   // (512,1024) fp32
  float* out = (float*)d_out;                 // (8,8,128,128,49) fp32

  char* ws = (char*)d_ws;
  const size_t xh_bytes = (size_t)MTOT * DIM * sizeof(_Float16);   // 134 MB
  const size_t wt_bytes = (size_t)ND * DIM * sizeof(_Float16);     //   1 MB
  _Float16* xh = (_Float16*)ws;
  _Float16* wt = (_Float16*)(ws + xh_bytes);
  float*    qk = (float*)(ws + xh_bytes + wt_bytes);               // 536 MB

  const int nelem_x = MTOT * DIM;                                  // 67108864
  cvt_x_kernel<<<nelem_x / (256 * 8), 256, 0, stream>>>(x, xh, nelem_x);
  cvt_w_kernel<<<(ND / 16) * (DIM / 16), 256, 0, stream>>>(wqk, wt);
  gemm_qk_kernel<<<(MTOT / 128) * (ND / 128), 128, 0, stream>>>(xh, wt, qk);
  neigh_attn_kernel<<<MB * NHEAD * (HH / TS) * (WW / TS), 256, 0, stream>>>(qk, out);
}
